// MultiHeadAttention_62766652063880
// MI455X (gfx1250) — matrix operand order repaired and checked by reading, not compiled, not run on hardware
//
#include <hip/hip_runtime.h>

#define DEV __device__ __forceinline__

typedef __attribute__((ext_vector_type(16))) __bf16 v16bf;
typedef __attribute__((ext_vector_type(8)))  float  v8f;

union FragAB { unsigned u[8]; v16bf v; };

// ---- constants for this problem ----
#define BATCH   2
#define SEQ     2048
#define DMODEL  1024
#define NHEAD   16
#define DHEAD   64
#define BS      (BATCH * SEQ)          // 4096 rows
#define LDS_STRIDE 72                  // padded row stride for 64-wide K tiles

DEV unsigned short f32_to_bf16(float f) {
  unsigned u = __float_as_uint(f);
  unsigned r = u + 0x7FFFu + ((u >> 16) & 1u);   // round-to-nearest-even
  return (unsigned short)(r >> 16);
}

// ---------------------------------------------------------------------------
// CDNA5 async staging: GLOBAL_LOAD_ASYNC_TO_LDS_B128 (ASYNCcnt-tracked).
// ---------------------------------------------------------------------------
DEV void async_copy_b128(const void* gptr, void* lptr) {
  unsigned lds_off = (unsigned)(size_t)lptr;
  unsigned long long ga = (unsigned long long)(size_t)gptr;
  asm volatile("global_load_async_to_lds_b128 %0, %1, off"
               :: "v"(lds_off), "v"(ga)
               : "memory");
}

template <int N>
DEV void wait_async() {
  asm volatile("s_wait_asynccnt %0" :: "i"(N) : "memory");
}

// Stage 128x64 bf16 tile (row-major, ld elems): 4 async B128 per thread.
DEV void stageA64_async(unsigned short* lds, const unsigned short* g, int ld, int tid) {
#pragma unroll
  for (int i = 0; i < 4; ++i) {
    int id  = tid + i * 256;
    int row = id >> 3;
    int c8  = (id & 7) * 8;
    async_copy_b128(g + (size_t)row * ld + c8, lds + row * LDS_STRIDE + c8);
  }
}

// Stage 64x64 bf16 tile: 2 async B128 per thread.
DEV void stageB64_async(unsigned short* lds, const unsigned short* g, int ld, int tid) {
#pragma unroll
  for (int i = 0; i < 2; ++i) {
    int id  = tid + i * 256;
    int row = id >> 3;
    int c8  = (id & 7) * 8;
    async_copy_b128(g + (size_t)row * ld + c8, lds + row * LDS_STRIDE + c8);
  }
}

// Stage 128x64 tile from f32 global, converting to bf16 (sync path).
DEV void stageA64_f32(unsigned short* lds, const float* g, int ld, int tid) {
#pragma unroll
  for (int i = 0; i < 4; ++i) {
    int id  = tid + i * 256;
    int row = id >> 3;
    int c8  = (id & 7) * 8;
    const float4* s4 = (const float4*)(g + (size_t)row * ld + c8);
    float4 x = s4[0];
    float4 y = s4[1];
    unsigned short* d = lds + row * LDS_STRIDE + c8;
    d[0] = f32_to_bf16(x.x); d[1] = f32_to_bf16(x.y);
    d[2] = f32_to_bf16(x.z); d[3] = f32_to_bf16(x.w);
    d[4] = f32_to_bf16(y.x); d[5] = f32_to_bf16(y.y);
    d[6] = f32_to_bf16(y.z); d[7] = f32_to_bf16(y.w);
  }
}

// ---------------------------------------------------------------------------
// WMMA fragments (CDNA5 ISA layouts).  The lds pointer may be
// pre-offset by a 32-column K-subtile (rows have stride LDS_STRIDE).
// ---------------------------------------------------------------------------
// A 16x32 bf16: lane&15 = row M, lane>>4 selects K octet pairs at +0/+16.
DEV void load_fragA(FragAB& a, const unsigned short* lds, int mbase, int lane) {
  const unsigned short* p = lds + (mbase + (lane & 15)) * LDS_STRIDE + ((lane >> 4) * 8);
#pragma unroll
  for (int j = 0; j < 4; ++j) {
    a.u[j]     = *(const unsigned*)(p + 2 * j);
    a.u[4 + j] = *(const unsigned*)(p + 16 + 2 * j);
  }
}

// B 32x16 bf16, operand stored K-contiguous per column (Bt[N][K]).
DEV void load_fragB(FragAB& b, const unsigned short* lds, int nbase, int lane) {
  const unsigned short* p = lds + (nbase + (lane & 15)) * LDS_STRIDE + ((lane >> 4) * 8);
#pragma unroll
  for (int j = 0; j < 4; ++j) {
    b.u[j]     = *(const unsigned*)(p + 2 * j);
    b.u[4 + j] = *(const unsigned*)(p + 16 + 2 * j);
  }
}

// Wave computes a 32x32 C-tile: 2 A-frags x 2 B-frags -> 4 WMMAs per 32-K sub.
DEV void wave_mma32(const unsigned short* lA, const unsigned short* lB,
                    int wm, int wn, int lane, v8f acc[2][2]) {
  FragAB a0, a1, b0, b1;
  load_fragA(a0, lA, wm,      lane);
  load_fragA(a1, lA, wm + 16, lane);
  load_fragB(b0, lB, wn,      lane);
  load_fragB(b1, lB, wn + 16, lane);
  acc[0][0] = __builtin_amdgcn_wmma_f32_16x16x32_bf16(false, a0.v, false, b0.v,
                                                      (short)0, acc[0][0], false, false);
  acc[0][1] = __builtin_amdgcn_wmma_f32_16x16x32_bf16(false, a0.v, false, b1.v,
                                                      (short)0, acc[0][1], false, false);
  acc[1][0] = __builtin_amdgcn_wmma_f32_16x16x32_bf16(false, a1.v, false, b0.v,
                                                      (short)0, acc[1][0], false, false);
  acc[1][1] = __builtin_amdgcn_wmma_f32_16x16x32_bf16(false, a1.v, false, b1.v,
                                                      (short)0, acc[1][1], false, false);
}

// 64-wide K step = two 32-K subtiles = 8 WMMAs.
DEV void wave_mma64(const unsigned short* lA, const unsigned short* lB,
                    int wm, int wn, int lane, v8f acc[2][2]) {
  wave_mma32(lA,      lB,      wm, wn, lane, acc);
  wave_mma32(lA + 32, lB + 32, wm, wn, lane, acc);
}

// ---------------------------------------------------------------------------
// Elementwise converters
// ---------------------------------------------------------------------------
__global__ __launch_bounds__(256) void cvt_bf16_kernel(const float* __restrict__ in,
                                                       unsigned short* __restrict__ out,
                                                       size_t n) {
  size_t i = (size_t)blockIdx.x * blockDim.x + threadIdx.x;
  size_t stride = (size_t)gridDim.x * blockDim.x;
  for (; i < n; i += stride) out[i] = f32_to_bf16(in[i]);
}

// W [K=1024][N=1024] f32  ->  WT [N][K] bf16
__global__ __launch_bounds__(256) void transpose_cvt_kernel(const float* __restrict__ W,
                                                            unsigned short* __restrict__ WT) {
  size_t i = (size_t)blockIdx.x * blockDim.x + threadIdx.x;
  int k = (int)(i >> 10);
  int n = (int)(i & 1023);
  WT[(size_t)n * DMODEL + k] = f32_to_bf16(W[(size_t)k * DMODEL + n]);
}

// ---------------------------------------------------------------------------
// WMMA GEMM: C[M,N] = A[M,K] * Bt[N,K]^T + bias[N]
//   mode 0: f32 [M][N]   mode 1: bf16 [b][h][s][d]   mode 2: bf16 [b][h][d][s]
// grid: (N/64, M/128); 256 threads, 8 waves of 32x32 C each.
// Double-buffered async-to-LDS pipeline, K-step 64, peeled last step.
// ---------------------------------------------------------------------------
__global__ __launch_bounds__(256)
void gemm_bf16_kernel(const unsigned short* __restrict__ A,
                      const unsigned short* __restrict__ Bt,
                      const float* __restrict__ bias,
                      void* __restrict__ out,
                      int N, int K, int mode) {
  __shared__ unsigned short lA[2][128 * LDS_STRIDE];
  __shared__ unsigned short lB[2][64 * LDS_STRIDE];

  int tid  = threadIdx.x;
  int lane = tid & 31;
  int wave = tid >> 5;
  int wm   = (wave >> 1) * 32;   // 0,32,64,96
  int wn   = (wave & 1) * 32;    // 0,32
  int m0   = blockIdx.y * 128;
  int n0   = blockIdx.x * 64;

  const unsigned short* Abase = A  + (size_t)m0 * K;
  const unsigned short* Bbase = Bt + (size_t)n0 * K;

  v8f acc[2][2] = {};

  stageA64_async(lA[0], Abase, K, tid);
  stageB64_async(lB[0], Bbase, K, tid);

  int nsteps = K >> 6;
  for (int it = 0; it < nsteps - 1; ++it) {
    int cur = it & 1;
    stageA64_async(lA[cur ^ 1], Abase + (it + 1) * 64, K, tid);
    stageB64_async(lB[cur ^ 1], Bbase + (it + 1) * 64, K, tid);
    wait_async<6>();            // allow the 6 just-issued; current tile landed
    __syncthreads();
    wave_mma64(lA[cur], lB[cur], wm, wn, lane, acc);
    __syncthreads();            // all reads done before next overwrite
  }
  wait_async<0>();
  __syncthreads();
  {
    int cur = (nsteps - 1) & 1;
    wave_mma64(lA[cur], lB[cur], wm, wn, lane, acc);
  }

#pragma unroll
  for (int ai = 0; ai < 2; ++ai) {
#pragma unroll
    for (int bi = 0; bi < 2; ++bi) {
      int n    = n0 + wn + bi * 16 + (lane & 15);
      int mrow = m0 + wm + ai * 16 + ((lane >> 4) * 8);
      float bn = bias[n];
#pragma unroll
      for (int r = 0; r < 8; ++r) {
        int m = mrow + r;
        float val = acc[ai][bi][r] + bn;
        if (mode == 0) {
          ((float*)out)[(size_t)m * N + n] = val;
        } else {
          unsigned short* O = (unsigned short*)out;
          int b = m >> 11, s = m & 2047;
          int h = n >> 6,  d = n & 63;
          if (mode == 1)
            O[((size_t)(b * NHEAD + h) * SEQ + s) * DHEAD + d] = f32_to_bf16(val);
          else
            O[((size_t)(b * NHEAD + h) * DHEAD + d) * SEQ + s] = f32_to_bf16(val);
        }
      }
    }
  }
}

// ---------------------------------------------------------------------------
// Logits: attn[bh][s][t] = 0.125 * sum_d qh[bh][s][d]*kh[bh][t][d] - 1e9*mask[b][t]
// grid: (SEQ/64, SEQ/128, B*H); K = 64 -> single-shot stage, 8 WMMAs.
// ---------------------------------------------------------------------------
__global__ __launch_bounds__(256)
void logits_kernel(const unsigned short* __restrict__ qh,
                   const unsigned short* __restrict__ kh,
                   const float* __restrict__ mask,
                   float* __restrict__ attn) {
  __shared__ unsigned short lA[128 * LDS_STRIDE];
  __shared__ unsigned short lB[64 * LDS_STRIDE];

  int tid  = threadIdx.x;
  int lane = tid & 31;
  int wave = tid >> 5;
  int wm   = (wave >> 1) * 32;
  int wn   = (wave & 1) * 32;
  int bh   = blockIdx.z;
  int b    = bh >> 4;
  int m0   = blockIdx.y * 128;   // s
  int n0   = blockIdx.x * 64;    // t

  const unsigned short* Abase = qh + (size_t)bh * SEQ * DHEAD + (size_t)m0 * DHEAD;
  const unsigned short* Bbase = kh + (size_t)bh * SEQ * DHEAD + (size_t)n0 * DHEAD;

  v8f acc[2][2] = {};

  stageA64_async(lA, Abase, DHEAD, tid);
  stageB64_async(lB, Bbase, DHEAD, tid);
  wait_async<0>();
  __syncthreads();
  wave_mma64(lA, lB, wm, wn, lane, acc);

  float* row_base = attn + (size_t)bh * SEQ * SEQ;
#pragma unroll
  for (int ai = 0; ai < 2; ++ai) {
#pragma unroll
    for (int bi = 0; bi < 2; ++bi) {
      int t    = n0 + wn + bi * 16 + (lane & 15);
      int srow = m0 + wm + ai * 16 + ((lane >> 4) * 8);
      float mterm = mask[b * SEQ + t] * -1e9f;
#pragma unroll
      for (int r = 0; r < 8; ++r)
        row_base[(size_t)(srow + r) * SEQ + t] = acc[ai][bi][r] * 0.125f + mterm;
    }
  }
}

// ---------------------------------------------------------------------------
// In-place row softmax (2048 wide). Wave32 shuffle reductions.
// ---------------------------------------------------------------------------
__global__ __launch_bounds__(256)
void softmax_kernel(float* __restrict__ attn) {
  __shared__ float red[8];
  float* p = attn + (size_t)blockIdx.x * SEQ;
  int tid  = threadIdx.x;
  int lane = tid & 31;
  int wv   = tid >> 5;

  float vals[8];
  float mx = -3.0e38f;
#pragma unroll
  for (int i = 0; i < 8; ++i) {
    vals[i] = p[tid + i * 256];
    mx = fmaxf(mx, vals[i]);
  }
#pragma unroll
  for (int o = 16; o > 0; o >>= 1) mx = fmaxf(mx, __shfl_xor(mx, o, 32));
  if (lane == 0) red[wv] = mx;
  __syncthreads();
  mx = red[0];
#pragma unroll
  for (int j = 1; j < 8; ++j) mx = fmaxf(mx, red[j]);
  __syncthreads();

  float sum = 0.f;
#pragma unroll
  for (int i = 0; i < 8; ++i) {
    vals[i] = __expf(vals[i] - mx);
    sum += vals[i];
  }
#pragma unroll
  for (int o = 16; o > 0; o >>= 1) sum += __shfl_xor(sum, o, 32);
  if (lane == 0) red[wv] = sum;
  __syncthreads();
  sum = red[0];
#pragma unroll
  for (int j = 1; j < 8; ++j) sum += red[j];

  float inv = 1.0f / sum;
#pragma unroll
  for (int i = 0; i < 8; ++i) p[tid + i * 256] = vals[i] * inv;
}

// ---------------------------------------------------------------------------
// ctx = attn @ V.  attn f32 [bh][s][t], vhT bf16 [bh][d][t].
// Output bf16 ctx in [b][s][h][d] (= [B][S][DMODEL]).
// grid: (1, SEQ/128, B*H); K = 2048. A staged sync (f32->bf16), B async.
// ---------------------------------------------------------------------------
__global__ __launch_bounds__(256)
void ctx_kernel(const float* __restrict__ attn,
                const unsigned short* __restrict__ vhT,
                unsigned short* __restrict__ ctxB) {
  __shared__ unsigned short lA[2][128 * LDS_STRIDE];
  __shared__ unsigned short lB[2][64 * LDS_STRIDE];

  int tid  = threadIdx.x;
  int lane = tid & 31;
  int wave = tid >> 5;
  int wm   = (wave >> 1) * 32;
  int wn   = (wave & 1) * 32;
  int bh   = blockIdx.z;
  int b    = bh >> 4;
  int h    = bh & 15;
  int m0   = blockIdx.y * 128;   // s

  const float* Abase          = attn + ((size_t)bh * SEQ + m0) * SEQ;
  const unsigned short* Bbase = vhT + (size_t)bh * DHEAD * SEQ;

  v8f acc[2][2] = {};

  stageA64_f32(lA[0], Abase, SEQ, tid);
  stageB64_async(lB[0], Bbase, SEQ, tid);

  const int nsteps = SEQ >> 6;   // 32
  for (int it = 0; it < nsteps - 1; ++it) {
    int cur = it & 1;
    stageA64_f32(lA[cur ^ 1], Abase + (it + 1) * 64, SEQ, tid);
    stageB64_async(lB[cur ^ 1], Bbase + (it + 1) * 64, SEQ, tid);
    wait_async<2>();
    __syncthreads();
    wave_mma64(lA[cur], lB[cur], wm, wn, lane, acc);
    __syncthreads();
  }
  wait_async<0>();
  __syncthreads();
  {
    int cur = (nsteps - 1) & 1;
    wave_mma64(lA[cur], lB[cur], wm, wn, lane, acc);
  }

#pragma unroll
  for (int ai = 0; ai < 2; ++ai) {
#pragma unroll
    for (int bi = 0; bi < 2; ++bi) {
      int d    = wn + bi * 16 + (lane & 15);
      int srow = m0 + wm + ai * 16 + ((lane >> 4) * 8);
#pragma unroll
      for (int r = 0; r < 8; ++r) {
        int s = srow + r;
        ctxB[((size_t)(b * SEQ + s) * NHEAD + h) * DHEAD + d] = f32_to_bf16(acc[ai][bi][r]);
      }
    }
  }
}

// ---------------------------------------------------------------------------
extern "C" void kernel_launch(void* const* d_in, const int* in_sizes, int n_in,
                              void* d_out, int out_size, void* d_ws, size_t ws_size,
                              hipStream_t stream) {
  (void)in_sizes; (void)n_in; (void)out_size; (void)ws_size;

  const float* q    = (const float*)d_in[0];
  const float* k    = (const float*)d_in[1];
  const float* v    = (const float*)d_in[2];
  const float* mask = (const float*)d_in[3];
  const float* wq   = (const float*)d_in[4];
  const float* bq   = (const float*)d_in[5];
  const float* wk   = (const float*)d_in[6];
  const float* bk   = (const float*)d_in[7];
  const float* wv   = (const float*)d_in[8];
  const float* bv   = (const float*)d_in[9];
  const float* wo   = (const float*)d_in[10];
  const float* bo   = (const float*)d_in[11];

  float* out  = (float*)d_out;
  float* attn = out + (size_t)BATCH * SEQ * DMODEL;   // +4,194,304

  const size_t WSZ = (size_t)DMODEL * DMODEL;   // 1M elems
  const size_t XSZ = (size_t)BS * DMODEL;       // 4M elems
  unsigned short* ws   = (unsigned short*)d_ws;
  unsigned short* WTq  = ws;
  unsigned short* WTk  = WTq + WSZ;
  unsigned short* WTv  = WTk + WSZ;
  unsigned short* WTo  = WTv + WSZ;
  unsigned short* qb   = WTo + WSZ;
  unsigned short* kb   = qb + XSZ;
  unsigned short* vb   = kb + XSZ;
  unsigned short* qh   = vb + XSZ;
  unsigned short* kh   = qh + XSZ;
  unsigned short* vhT  = kh + XSZ;
  unsigned short* ctxB = vhT + XSZ;

  // 1) transpose+convert weights to bf16 [N][K]
  transpose_cvt_kernel<<<4096, 256, 0, stream>>>(wq, WTq);
  transpose_cvt_kernel<<<4096, 256, 0, stream>>>(wk, WTk);
  transpose_cvt_kernel<<<4096, 256, 0, stream>>>(wv, WTv);
  transpose_cvt_kernel<<<4096, 256, 0, stream>>>(wo, WTo);

  // 2) convert activations to bf16
  cvt_bf16_kernel<<<2048, 256, 0, stream>>>(q, qb, XSZ);
  cvt_bf16_kernel<<<2048, 256, 0, stream>>>(k, kb, XSZ);
  cvt_bf16_kernel<<<2048, 256, 0, stream>>>(v, vb, XSZ);

  // 3) projections (WMMA), head layouts (V transposed)
  dim3 gproj(DMODEL / 64, BS / 128);
  gemm_bf16_kernel<<<gproj, 256, 0, stream>>>(qb, WTq, bq, qh,  DMODEL, DMODEL, 1);
  gemm_bf16_kernel<<<gproj, 256, 0, stream>>>(kb, WTk, bk, kh,  DMODEL, DMODEL, 1);
  gemm_bf16_kernel<<<gproj, 256, 0, stream>>>(vb, WTv, bv, vhT, DMODEL, DMODEL, 2);

  // 4) logits into attn region of d_out
  dim3 glog(SEQ / 64, SEQ / 128, BATCH * NHEAD);
  logits_kernel<<<glog, 256, 0, stream>>>(qh, kh, mask, attn);

  // 5) softmax in place
  softmax_kernel<<<BATCH * NHEAD * SEQ, 256, 0, stream>>>(attn);

  // 6) ctx = attn @ V
  dim3 gctx(1, SEQ / 128, BATCH * NHEAD);
  ctx_kernel<<<gctx, 256, 0, stream>>>(attn, vhT, ctxB);

  // 7) out = ctx @ Wo + bo
  gemm_bf16_kernel<<<gproj, 256, 0, stream>>>(ctxB, WTo, bo, out, DMODEL, DMODEL, 0);
}
